// PointConvUp_44693429682657
// MI455X (gfx1250) — compile-verified, matrix operand order checked by reading, not run on hardware
//
#include <hip/hip_runtime.h>
#include <hip/hip_bf16.h>
#include <stdint.h>

// ---------------- problem constants (match reference setup_inputs) ----------
#define B_     8
#define N1_    4096
#define N2_    1024
#define D_     64
#define K_     16
#define H_     128
#define HU_    8
#define CIN_   131     // 64 + 64 + 3
#define CINP_  160     // padded to 5 * 32 for bf16 WMMA K-chunks
#define LEAKY_ 0.1f

typedef __attribute__((ext_vector_type(16))) __bf16 v16bf;
typedef __attribute__((ext_vector_type(8)))  float  v8f;

// ---------------- helpers ---------------------------------------------------
__device__ __forceinline__ unsigned short bf16bits(float f) {
  unsigned u = __builtin_bit_cast(unsigned, f);
  u += 0x7fffu + ((u >> 16) & 1u);            // round-to-nearest-even
  return (unsigned short)(u >> 16);
}
__device__ __forceinline__ unsigned pack_bf2(float a, float b) {
  return (unsigned)bf16bits(a) | ((unsigned)bf16bits(b) << 16);
}
__device__ __forceinline__ void lds_fence() {
  // CDNA5 split dependency counter: wait for all outstanding LDS ops of this wave
  asm volatile("s_wait_dscnt 0" ::: "memory");
}
__device__ __forceinline__ float leaky(float x) { return x > 0.f ? x : LEAKY_ * x; }

// A-matrix fragment (16x32 bf16, ISA 7.12.2 layout) gathered from an LDS
// row-major tile  [16 rows][stride elems].
__device__ __forceinline__ v16bf load_a_frag(const __bf16* s, int kcbase, int stride) {
  const int lane = threadIdx.x & 31;
  const int m    = lane & 15;
  const int kb   = (lane >> 4) << 3;          // lanes 16..31 hold K+8 halves
  const unsigned short* row = (const unsigned short*)s + m * stride + kcbase + kb;
  union { unsigned u[8]; v16bf v; } r;
  r.u[0] = *(const unsigned*)(row + 0);       // K = 0,1
  r.u[1] = *(const unsigned*)(row + 2);       // K = 2,3
  r.u[2] = *(const unsigned*)(row + 4);       // K = 4,5
  r.u[3] = *(const unsigned*)(row + 6);       // K = 6,7
  r.u[4] = *(const unsigned*)(row + 16);      // K = 16,17
  r.u[5] = *(const unsigned*)(row + 18);      // K = 18,19
  r.u[6] = *(const unsigned*)(row + 20);      // K = 20,21
  r.u[7] = *(const unsigned*)(row + 22);      // K = 22,23
  return r.v;
}

// B-matrix fragment pre-packed by the prep kernels: each lane owns 32
// contiguous bytes -> two 16B global loads, always hot in L2.
__device__ __forceinline__ v16bf load_b_frag(const uint4* base, int fid, int lane) {
  union { uint4 q[2]; v16bf v; } r;
  r.q[0] = base[(size_t)(fid * 32 + lane) * 2 + 0];
  r.q[1] = base[(size_t)(fid * 32 + lane) * 2 + 1];
  return r.v;
}

// ---------------- prep: fold BN into conv weights, pack bf16 fragments ------
__global__ void prep_scale_kernel(
    const float* __restrict__ c0_b, const float* __restrict__ g0,
    const float* __restrict__ be0, const float* __restrict__ m0, const float* __restrict__ v0,
    const float* __restrict__ c1_b, const float* __restrict__ g1,
    const float* __restrict__ be1, const float* __restrict__ m1, const float* __restrict__ v1,
    float* __restrict__ b0f, float* __restrict__ b1f,
    float* __restrict__ s0f, float* __restrict__ s1f) {
  int n = blockIdx.x * blockDim.x + threadIdx.x;
  if (n >= H_) return;
  float s0 = g0[n] * rsqrtf(v0[n] + 1e-5f);
  s0f[n] = s0;
  b0f[n] = (c0_b[n] - m0[n]) * s0 + be0[n];
  float s1 = g1[n] * rsqrtf(v1[n] + 1e-5f);
  s1f[n] = s1;
  b1f[n] = (c1_b[n] - m1[n]) * s1 + be1[n];
}

template<int CIN, int NCHUNK>
__global__ void prep_pack_kernel(const float* __restrict__ w, const float* __restrict__ scale,
                                 __bf16* __restrict__ pack) {
  int t = blockIdx.x * blockDim.x + threadIdx.x;     // one thread = one lane of one fragment
  if (t >= NCHUNK * 8 * 32) return;
  int fid   = t >> 5;
  int lane  = t & 31;
  int chunk = fid >> 3;
  int nt    = fid & 7;
  int n     = nt * 16 + (lane & 15);
  int kb    = (lane >> 4) << 3;
  union { __bf16 h[16]; uint4 q[2]; } u;
#pragma unroll
  for (int i = 0; i < 16; ++i) {
    int k = chunk * 32 + ((i < 8) ? (i + kb) : (16 + (i - 8) + kb));
    float v = (k < CIN) ? w[(size_t)n * CIN + k] * scale[n] : 0.f;
    u.h[i] = __builtin_bit_cast(__bf16, bf16bits(v));
  }
  uint4* dst = (uint4*)pack + (size_t)t * 2;
  dst[0] = u.q[0];
  dst[1] = u.q[1];
}

// ---------------- brute-force KNN (top-16) ----------------------------------
template<int NREF>
__global__ __launch_bounds__(256) void knn_kernel(const float* __restrict__ qxyz,
                                                  const float* __restrict__ rxyz,
                                                  int* __restrict__ idx) {
  __shared__ float sx[NREF], sy[NREF], sz[NREF];
  const int b = blockIdx.y;
  const float* rb = rxyz + (size_t)b * 3 * NREF;
  for (int t = threadIdx.x; t < NREF; t += 256) {
    sx[t] = rb[t];
    sy[t] = rb[NREF + t];
    sz[t] = rb[2 * NREF + t];
  }
  __syncthreads();

  const int q = blockIdx.x * 256 + threadIdx.x;
  const float* qb = qxyz + (size_t)b * 3 * N1_;
  const float qx = qb[q], qy = qb[N1_ + q], qz = qb[2 * N1_ + q];

  float bd[K_]; int bi[K_];
#pragma unroll
  for (int i = 0; i < K_; ++i) { bd[i] = 3.4e38f; bi[i] = 0; }

  for (int r = 0; r < NREF; ++r) {
    float dx = sx[r] - qx, dy = sy[r] - qy, dz = sz[r] - qz;
    float d = dx * dx + dy * dy + dz * dz;
    if (d < bd[K_ - 1]) {
      float vd = d; int vi = r;
#pragma unroll
      for (int i = 0; i < K_; ++i) {            // register insertion sort
        bool c = vd < bd[i];
        float td = bd[i]; int ti = bi[i];
        bd[i] = c ? vd : td; bi[i] = c ? vi : ti;
        vd    = c ? td : vd; vi    = c ? ti : vi;
      }
    }
  }
  int* o = idx + (((size_t)b * N1_) + q) * K_;
#pragma unroll
  for (int i = 0; i < K_; ++i) o[i] = bi[i];
}

// ---------------- stage 1: fused grouping + conv0/conv1 (WMMA) + weightnet --
#define WPB1 4   // waves per block, one query per wave
__global__ __launch_bounds__(WPB1 * 32) void stage1_kernel(
    const float* __restrict__ xyz1, const float* __restrict__ xyz2,
    const float* __restrict__ points1, const float* __restrict__ points2,
    const int* __restrict__ idx1,
    const __bf16* __restrict__ w0pack, const __bf16* __restrict__ w1pack,
    const float* __restrict__ b0f, const float* __restrict__ b1f,
    const float* __restrict__ wn0w, const float* __restrict__ wn0b,
    const float* __restrict__ wn1w, const float* __restrict__ wn1b,
    const float* __restrict__ wn2w, const float* __restrict__ wn2b,
    float* __restrict__ cost) {
  __shared__ __bf16 stage[WPB1][16][CINP_];   // 20 KB: per-wave A tile
  __shared__ float  hwn[WPB1][16][HU_];       // weightnet hidden per neighbor row
  __shared__ float  sdir[WPB1][16][4];

  const int lane = threadIdx.x & 31;
  const int wv   = threadIdx.x >> 5;
  const int b    = blockIdx.y;
  const int q    = blockIdx.x * WPB1 + wv;
  const int m    = lane & 15;                 // neighbor row / out-col low bits
  const int hi   = lane >> 4;

  // ---- build A tile: [16 rows(k) x 160] = concat(f1, f2[idx], dir, pad) ----
  {  // f1 broadcast: each lane owns 2 channels, writes all 16 rows
    int c = lane * 2;
    float v0 = points1[((size_t)b * D_ + c) * N1_ + q];
    float v1 = points1[((size_t)b * D_ + c + 1) * N1_ + q];
    unsigned pk = pack_bf2(v0, v1);
#pragma unroll
    for (int r = 0; r < 16; ++r) *(unsigned*)&stage[wv][r][c] = pk;
  }
  const int j = idx1[(((size_t)b * N1_) + q) * K_ + m];
  {  // gathered f2: row m, channel half hi
    int cb = hi * 32;
#pragma unroll
    for (int t = 0; t < 32; t += 2) {
      float v0 = points2[((size_t)b * D_ + cb + t) * N2_ + j];
      float v1 = points2[((size_t)b * D_ + cb + t + 1) * N2_ + j];
      *(unsigned*)&stage[wv][m][D_ + cb + t] = pack_bf2(v0, v1);
    }
  }
  if (hi == 0) {  // dir_xyz (cols 128..130) + one pad col
    float dx = xyz2[((size_t)b * 3 + 0) * N2_ + j] - xyz1[((size_t)b * 3 + 0) * N1_ + q];
    float dy = xyz2[((size_t)b * 3 + 1) * N2_ + j] - xyz1[((size_t)b * 3 + 1) * N1_ + q];
    float dz = xyz2[((size_t)b * 3 + 2) * N2_ + j] - xyz1[((size_t)b * 3 + 2) * N1_ + q];
    sdir[wv][m][0] = dx; sdir[wv][m][1] = dy; sdir[wv][m][2] = dz;
    *(unsigned*)&stage[wv][m][128] = pack_bf2(dx, dy);
    *(unsigned*)&stage[wv][m][130] = pack_bf2(dz, 0.f);
  } else {        // zero padding cols 132..159
#pragma unroll
    for (int t = 132; t < CINP_; t += 2) *(unsigned*)&stage[wv][m][t] = 0u;
  }
  lds_fence();

  // ---- WeightNet1 hidden (3 -> 8 -> 8, ReLU), one neighbor row per lane ----
  if (hi == 0) {
    float d0 = sdir[wv][m][0], d1 = sdir[wv][m][1], d2 = sdir[wv][m][2];
    float h1[HU_];
#pragma unroll
    for (int u = 0; u < HU_; ++u)
      h1[u] = fmaxf(wn0w[u * 3 + 0] * d0 + wn0w[u * 3 + 1] * d1 + wn0w[u * 3 + 2] * d2 + wn0b[u], 0.f);
#pragma unroll
    for (int u = 0; u < HU_; ++u) {
      float s = wn1b[u];
#pragma unroll
      for (int t = 0; t < HU_; ++t) s += wn1w[u * HU_ + t] * h1[t];
      hwn[wv][m][u] = fmaxf(s, 0.f);
    }
  }

  // ---- GEMM1: [16 x 160] x [160 x 128] in bf16 WMMA, f32 accumulate --------
  v8f acc[8];
#pragma unroll
  for (int nt = 0; nt < 8; ++nt) acc[nt] = (v8f){0.f, 0.f, 0.f, 0.f, 0.f, 0.f, 0.f, 0.f};
  const uint4* w0q = (const uint4*)w0pack;
#pragma unroll
  for (int kc = 0; kc < 5; ++kc) {
    v16bf a = load_a_frag(&stage[wv][0][0], kc * 32, CINP_);
#pragma unroll
    for (int nt = 0; nt < 8; ++nt) {
      v16bf bf = load_b_frag(w0q, kc * 8 + nt, lane);
      acc[nt] = __builtin_amdgcn_wmma_f32_16x16x32_bf16(false, a, false, bf,
                                                        (short)0, acc[nt], false, false);
    }
  }
  lds_fence();

  // ---- bias0 (BN folded) + leaky, restage H1 as bf16 [16 x 128] ------------
#pragma unroll
  for (int nt = 0; nt < 8; ++nt) {
    float bb = b0f[nt * 16 + m];
#pragma unroll
    for (int r = 0; r < 8; ++r) {
      float x = leaky(acc[nt][r] + bb);
      stage[wv][r + hi * 8][nt * 16 + m] = __builtin_bit_cast(__bf16, bf16bits(x));
    }
  }
  lds_fence();

  // ---- GEMM2: [16 x 128] x [128 x 128] -------------------------------------
  v8f acc2[8];
#pragma unroll
  for (int nt = 0; nt < 8; ++nt) acc2[nt] = (v8f){0.f, 0.f, 0.f, 0.f, 0.f, 0.f, 0.f, 0.f};
  const uint4* w1q = (const uint4*)w1pack;
#pragma unroll
  for (int kc = 0; kc < 4; ++kc) {
    v16bf a = load_a_frag(&stage[wv][0][0], kc * 32, CINP_);
#pragma unroll
    for (int nt = 0; nt < 8; ++nt) {
      v16bf bf = load_b_frag(w1q, kc * 8 + nt, lane);
      acc2[nt] = __builtin_amdgcn_wmma_f32_16x16x32_bf16(false, a, false, bf,
                                                         (short)0, acc2[nt], false, false);
    }
  }

  // ---- bias1 + leaky, WeightNet final layer on the fly, reduce over K ------
#pragma unroll
  for (int nt = 0; nt < 8; ++nt) {
    int   c  = nt * 16 + m;
    float bb = b1f[c];
    float wb = wn2b[c];
    float w8[HU_];
#pragma unroll
    for (int t = 0; t < HU_; ++t) w8[t] = wn2w[c * HU_ + t];
    float s = 0.f;
#pragma unroll
    for (int r = 0; r < 8; ++r) {
      int mm = r + hi * 8;
      float h2v = leaky(acc2[nt][r] + bb);
      float wvv = wb;
#pragma unroll
      for (int t = 0; t < HU_; ++t) wvv += hwn[wv][mm][t] * w8[t];
      wvv = fmaxf(wvv, 0.f);
      s += wvv * h2v;
    }
    s += __shfl_xor(s, 16, 32);   // combine the two 8-neighbor halves of the wave
    if (hi == 0) cost[(((size_t)b * N1_) + q) * H_ + c] = s;
  }
}

// ---------------- stage 2: self-KNN weighted smoothing ----------------------
#define WPB2 8
__global__ __launch_bounds__(WPB2 * 32) void stage2_kernel(
    const float* __restrict__ xyz1, const int* __restrict__ idx2,
    const float* __restrict__ cost,
    const float* __restrict__ wn0w, const float* __restrict__ wn0b,
    const float* __restrict__ wn1w, const float* __restrict__ wn1b,
    const float* __restrict__ wn2w, const float* __restrict__ wn2b,
    float* __restrict__ out) {
  __shared__ float hwn[WPB2][16][HU_];
  __shared__ int   sidx[WPB2][16];

  const int lane = threadIdx.x & 31;
  const int wv   = threadIdx.x >> 5;
  const int b    = blockIdx.y;
  const int q    = blockIdx.x * WPB2 + wv;
  const int m    = lane & 15;
  const int hi   = lane >> 4;

  if (hi == 0) {
    int j = idx2[(((size_t)b * N1_) + q) * K_ + m];
    sidx[wv][m] = j;
    float dx = xyz1[((size_t)b * 3 + 0) * N1_ + j] - xyz1[((size_t)b * 3 + 0) * N1_ + q];
    float dy = xyz1[((size_t)b * 3 + 1) * N1_ + j] - xyz1[((size_t)b * 3 + 1) * N1_ + q];
    float dz = xyz1[((size_t)b * 3 + 2) * N1_ + j] - xyz1[((size_t)b * 3 + 2) * N1_ + q];
    float h1[HU_];
#pragma unroll
    for (int u = 0; u < HU_; ++u)
      h1[u] = fmaxf(wn0w[u * 3 + 0] * dx + wn0w[u * 3 + 1] * dy + wn0w[u * 3 + 2] * dz + wn0b[u], 0.f);
#pragma unroll
    for (int u = 0; u < HU_; ++u) {
      float s = wn1b[u];
#pragma unroll
      for (int t = 0; t < HU_; ++t) s += wn1w[u * HU_ + t] * h1[t];
      hwn[wv][m][u] = fmaxf(s, 0.f);
    }
  }
  lds_fence();

  const float* costb = cost + (size_t)b * N1_ * H_;
#pragma unroll
  for (int nt = 0; nt < 8; ++nt) {
    int   c  = nt * 16 + m;
    float wb = wn2b[c];
    float w8[HU_];
#pragma unroll
    for (int t = 0; t < HU_; ++t) w8[t] = wn2w[c * HU_ + t];
    float s = 0.f;
#pragma unroll
    for (int r = 0; r < 8; ++r) {
      int mm = r + hi * 8;
      int j  = sidx[wv][mm];
      float wvv = wb;
#pragma unroll
      for (int t = 0; t < HU_; ++t) wvv += hwn[wv][mm][t] * w8[t];
      wvv = fmaxf(wvv, 0.f);
      s += wvv * costb[(size_t)j * H_ + c];
    }
    s += __shfl_xor(s, 16, 32);
    if (hi == 0) out[(((size_t)b * H_) + c) * N1_ + q] = s;   // [B, Cout, N1]
  }
}

// ---------------- launcher ---------------------------------------------------
extern "C" void kernel_launch(void* const* d_in, const int* in_sizes, int n_in,
                              void* d_out, int out_size, void* d_ws, size_t ws_size,
                              hipStream_t stream) {
  (void)in_sizes; (void)n_in; (void)out_size; (void)ws_size;
  const float* xyz1    = (const float*)d_in[0];
  const float* xyz2    = (const float*)d_in[1];
  const float* points1 = (const float*)d_in[2];
  const float* points2 = (const float*)d_in[3];
  const float* c0_w    = (const float*)d_in[4];
  const float* c0_b    = (const float*)d_in[5];
  const float* bn0_g   = (const float*)d_in[6];
  const float* bn0_b   = (const float*)d_in[7];
  const float* bn0_m   = (const float*)d_in[8];
  const float* bn0_v   = (const float*)d_in[9];
  const float* c1_w    = (const float*)d_in[10];
  const float* c1_b    = (const float*)d_in[11];
  const float* bn1_g   = (const float*)d_in[12];
  const float* bn1_b   = (const float*)d_in[13];
  const float* bn1_m   = (const float*)d_in[14];
  const float* bn1_v   = (const float*)d_in[15];
  const float* w1_0w   = (const float*)d_in[16];
  const float* w1_0b   = (const float*)d_in[17];
  const float* w1_1w   = (const float*)d_in[18];
  const float* w1_1b   = (const float*)d_in[19];
  const float* w1_2w   = (const float*)d_in[20];
  const float* w1_2b   = (const float*)d_in[21];
  const float* w2_0w   = (const float*)d_in[22];
  const float* w2_0b   = (const float*)d_in[23];
  const float* w2_1w   = (const float*)d_in[24];
  const float* w2_1b   = (const float*)d_in[25];
  const float* w2_2w   = (const float*)d_in[26];
  const float* w2_2b   = (const float*)d_in[27];
  float* out = (float*)d_out;

  char* ws = (char*)d_ws;
  size_t o = 0;
  int*    idx1   = (int*)(ws + o);    o += (size_t)B_ * N1_ * K_ * 4;
  int*    idx2   = (int*)(ws + o);    o += (size_t)B_ * N1_ * K_ * 4;
  float*  cost   = (float*)(ws + o);  o += (size_t)B_ * N1_ * H_ * 4;
  __bf16* w0pack = (__bf16*)(ws + o); o += (size_t)5 * 8 * 32 * 16 * 2;
  __bf16* w1pack = (__bf16*)(ws + o); o += (size_t)4 * 8 * 32 * 16 * 2;
  float*  b0f    = (float*)(ws + o);  o += H_ * 4;
  float*  b1f    = (float*)(ws + o);  o += H_ * 4;
  float*  s0f    = (float*)(ws + o);  o += H_ * 4;
  float*  s1f    = (float*)(ws + o);  o += H_ * 4;

  prep_scale_kernel<<<1, 128, 0, stream>>>(c0_b, bn0_g, bn0_b, bn0_m, bn0_v,
                                           c1_b, bn1_g, bn1_b, bn1_m, bn1_v,
                                           b0f, b1f, s0f, s1f);
  prep_pack_kernel<CIN_, 5><<<5, 256, 0, stream>>>(c0_w, s0f, w0pack);   // 1280 threads
  prep_pack_kernel<H_,   4><<<4, 256, 0, stream>>>(c1_w, s1f, w1pack);   // 1024 threads

  knn_kernel<N2_><<<dim3(N1_ / 256, B_), 256, 0, stream>>>(xyz1, xyz2, idx1);
  knn_kernel<N1_><<<dim3(N1_ / 256, B_), 256, 0, stream>>>(xyz1, xyz1, idx2);

  stage1_kernel<<<dim3(N1_ / WPB1, B_), WPB1 * 32, 0, stream>>>(
      xyz1, xyz2, points1, points2, idx1, w0pack, w1pack, b0f, b1f,
      w1_0w, w1_0b, w1_1w, w1_1b, w1_2w, w1_2b, cost);

  stage2_kernel<<<dim3(N1_ / WPB2, B_), WPB2 * 32, 0, stream>>>(
      xyz1, idx2, cost, w2_0w, w2_0b, w2_1w, w2_1b, w2_2w, w2_2b, out);
}
